// BornCollapseSampler_25142738551439
// MI455X (gfx1250) — matrix operand order repaired and checked by reading, not compiled, not run on hardware
//
#include <hip/hip_runtime.h>
#include <hip/hip_bf16.h>
#include <math.h>

// ---------------------------------------------------------------------------
// Problem constants (from reference): B=8, S=128, D=1024, V=32000
// ---------------------------------------------------------------------------
#define M_TOTAL 1024      // B*S
#define K_TOTAL 1024      // D
#define V_TOTAL 32000     // vocab
#define TOPK    50
#define TOPP    0.95f

#define TILE_M  64
#define TILE_N  128
#define KSTEP   32

typedef __attribute__((ext_vector_type(16))) __bf16 v16bf;
typedef __attribute__((ext_vector_type(8)))  float  v8f;

union Frag16 { v16bf v; uint4 q[2]; };   // 32 bytes = 16 bf16

__device__ __forceinline__ unsigned short f2bf_hw(float f) {
    union { __bf16 b; unsigned short s; } u;
    u.b = (__bf16)f;                      // hardware cvt
    return u.s;
}

// generic pointer -> 32-bit LDS byte offset (for async-to-LDS dest VGPR)
__device__ __forceinline__ unsigned lds_offset(const void* p) {
    return (unsigned)(unsigned long long)
        (__attribute__((address_space(3))) const void*)p;
}

// ---------------------------------------------------------------------------
// Kernel 0: fp32 -> bf16 pre-conversion (scale = +1 or -1), 8 elems/thread
// ---------------------------------------------------------------------------
__global__ __launch_bounds__(256)
void to_bf16_kernel(const float* __restrict__ in,
                    unsigned short* __restrict__ out, int n8, float scale) {
    int i = blockIdx.x * blockDim.x + threadIdx.x;
    if (i >= n8) return;
    const float4* p = reinterpret_cast<const float4*>(in) + (size_t)i * 2;
    float4 a = p[0], b = p[1];
    union { unsigned short s[8]; uint4 q; } o;
    o.s[0] = f2bf_hw(a.x * scale); o.s[1] = f2bf_hw(a.y * scale);
    o.s[2] = f2bf_hw(a.z * scale); o.s[3] = f2bf_hw(a.w * scale);
    o.s[4] = f2bf_hw(b.x * scale); o.s[5] = f2bf_hw(b.y * scale);
    o.s[6] = f2bf_hw(b.z * scale); o.s[7] = f2bf_hw(b.w * scale);
    reinterpret_cast<uint4*>(out)[i] = o.q;
}

// ---------------------------------------------------------------------------
// Kernel 1: complex projection via bf16 WMMA, double-buffered async staging.
//   acc_r = Ar*Br + (-Ai)*Bi ; acc_i = Ai*Br + Ar*Bi ; ampsq = r^2 + i^2
// Block: 256 threads = 8 waves; tile 64(M) x 128(N); wave w owns N sub-tile w.
// psi tiles staged with global_load_async_to_lds_b128; tile i+1 stays in
// flight (s_wait_asynccnt 3) while tile i feeds the 16 WMMAs.
// ---------------------------------------------------------------------------
__global__ __launch_bounds__(256)
void born_gemm_wmma(const unsigned short* __restrict__ prb,
                    const unsigned short* __restrict__ pib,
                    const unsigned short* __restrict__ pinb,
                    const unsigned short* __restrict__ wrb,
                    const unsigned short* __restrict__ wib,
                    float* __restrict__ ampsq_out) {
    __shared__ __align__(16) unsigned short sA[2][3][TILE_M][KSTEP];  // 24 KB

    const int tid   = threadIdx.x;
    const int lane  = tid & 31;
    const int wave  = tid >> 5;
    const int nlo   = lane & 15;
    const int khalf = lane >> 4;

    const int n_tile = blockIdx.x * TILE_N;
    const int m_tile = blockIdx.y * TILE_M;
    const int ncol   = n_tile + wave * 16 + nlo;

    const unsigned short* wr_row = wrb + (size_t)ncol * K_TOTAL;
    const unsigned short* wi_row = wib + (size_t)ncol * K_TOTAL;
    const unsigned short* mats[3] = { prb, pib, pinb };

    // this thread's 3 staging chunks (16B each), fixed across iterations
    int s_mat[3], s_row[3], s_c[3];
#pragma unroll
    for (int i = 0; i < 3; ++i) {
        int idx = tid + i * 256;
        s_mat[i] = idx >> 8;          // 256 chunks per matrix
        int rem  = idx & 255;
        s_row[i] = rem >> 2;
        s_c[i]   = rem & 3;           // 16B chunk within the 64B row
    }

    v8f acc_r[4], acc_i[4];
    const v8f vzero = {0.f, 0.f, 0.f, 0.f, 0.f, 0.f, 0.f, 0.f};
#pragma unroll
    for (int m = 0; m < 4; ++m) { acc_r[m] = vzero; acc_i[m] = vzero; }

    const int kb2 = khalf * 16;   // unused; B frag uses kb order like A
    const int kb  = khalf * 8;    // A frag: K kb..kb+7, kb+16..kb+23

    // ---- helpers -----------------------------------------------------------
    auto issue_stage = [&](int k0, int buf) {
#pragma unroll
        for (int i = 0; i < 3; ++i) {
            const unsigned short* src = mats[s_mat[i]]
                + (size_t)(m_tile + s_row[i]) * K_TOTAL + k0 + s_c[i] * 8;
            unsigned dst = lds_offset(&sA[buf][s_mat[i]][s_row[i]][s_c[i] * 8]);
            asm volatile("global_load_async_to_lds_b128 %0, %1, off"
                         :: "v"(dst), "v"((unsigned long long)(size_t)src)
                         : "memory");
        }
    };
    auto load_b = [&](int k0, Frag16& br, Frag16& bi) {
        const uint4* pr4 = reinterpret_cast<const uint4*>(wr_row + k0 + kb);
        const uint4* pi4 = reinterpret_cast<const uint4*>(wi_row + k0 + kb);
        br.q[0] = pr4[0]; br.q[1] = pr4[2];
        bi.q[0] = pi4[0]; bi.q[1] = pi4[2];
    };

    // ---- prologue: stage tile 0, load B fragments for k=0 ------------------
    issue_stage(0, 0);
    Frag16 brc, bic;
    load_b(0, brc, bic);

    const int NK = K_TOTAL / KSTEP;          // 32
    for (int it = 0; it < NK; ++it) {
        const int kcur  = it * KSTEP;
        const int knext = (kcur + KSTEP) & (K_TOTAL - 1);   // wraps on last
        const int bufc  = it & 1;
        const int bufn  = bufc ^ 1;

        // issue next tile's stage + next B loads; both overlap this compute
        issue_stage(knext, bufn);
        Frag16 brn, bin2;
        load_b(knext, brn, bin2);

        // wait only for the CURRENT tile's 3 async ops (next 3 stay in flight)
        asm volatile("s_wait_asynccnt 0x3" ::: "memory");
        __syncthreads();

        // ---- per M sub-tile: A frags from LDS (pure b128), 4 WMMAs ----
#pragma unroll
        for (int msub = 0; msub < 4; ++msub) {
            const int mrow = msub * 16 + nlo;
            Frag16 ar, ai, ain;
            const uint4* a0 = reinterpret_cast<const uint4*>(&sA[bufc][0][mrow][kb]);
            const uint4* a1 = reinterpret_cast<const uint4*>(&sA[bufc][1][mrow][kb]);
            const uint4* a2 = reinterpret_cast<const uint4*>(&sA[bufc][2][mrow][kb]);
            ar.q[0]  = a0[0]; ar.q[1]  = a0[2];   // +32B = K base + 16
            ai.q[0]  = a1[0]; ai.q[1]  = a1[2];
            ain.q[0] = a2[0]; ain.q[1] = a2[2];

            acc_r[msub] = __builtin_amdgcn_wmma_f32_16x16x32_bf16(
                false, ar.v,  false, brc.v, (short)0, acc_r[msub], false, false);
            acc_r[msub] = __builtin_amdgcn_wmma_f32_16x16x32_bf16(
                false, ain.v, false, bic.v, (short)0, acc_r[msub], false, false);
            acc_i[msub] = __builtin_amdgcn_wmma_f32_16x16x32_bf16(
                false, ai.v,  false, brc.v, (short)0, acc_i[msub], false, false);
            acc_i[msub] = __builtin_amdgcn_wmma_f32_16x16x32_bf16(
                false, ar.v,  false, bic.v, (short)0, acc_i[msub], false, false);
        }

        __syncthreads();      // everyone done reading bufc before it refills
        brc = brn; bic = bin2;
    }

    // ---- |amp|^2 (C layout: VGPR j -> M = j + 8*khalf, N = lane&15) ----
#pragma unroll
    for (int msub = 0; msub < 4; ++msub) {
#pragma unroll
        for (int j = 0; j < 8; ++j) {
            int m = m_tile + msub * 16 + j + khalf * 8;
            float cr = acc_r[msub][j];
            float ci = acc_i[msub][j];
            ampsq_out[(size_t)m * V_TOTAL + ncol] = cr * cr + ci * ci;
        }
    }
}

// ---------------------------------------------------------------------------
// Kernel 2: row-wise epilogue. One block per (b,s) row; full 32000-float row
// cached in dynamic LDS (128 KB of the WGP's 320 KB).
// ---------------------------------------------------------------------------
#define BLK 256

__device__ __forceinline__ float blk_sum(float v, float* red) {
    int t = threadIdx.x;
    __syncthreads(); red[t] = v; __syncthreads();
    for (int s = BLK / 2; s > 0; s >>= 1) {
        if (t < s) red[t] += red[t + s];
        __syncthreads();
    }
    return red[0];
}
__device__ __forceinline__ float blk_max(float v, float* red) {
    int t = threadIdx.x;
    __syncthreads(); red[t] = v; __syncthreads();
    for (int s = BLK / 2; s > 0; s >>= 1) {
        if (t < s) red[t] = fmaxf(red[t], red[t + s]);
        __syncthreads();
    }
    return red[0];
}
__device__ __forceinline__ float blk_min(float v, float* red) {
    int t = threadIdx.x;
    __syncthreads(); red[t] = v; __syncthreads();
    for (int s = BLK / 2; s > 0; s >>= 1) {
        if (t < s) red[t] = fminf(red[t], red[t + s]);
        __syncthreads();
    }
    return red[0];
}

__device__ __forceinline__ unsigned mix32(unsigned x) {
    x ^= x >> 16; x *= 0x7feb352du;
    x ^= x >> 15; x *= 0x846ca68bu;
    x ^= x >> 16; return x;
}

__global__ __launch_bounds__(BLK)
void born_rowwise(const float* __restrict__ ampsq,
                  const float* __restrict__ bias,
                  float* __restrict__ logits_out,
                  float* __restrict__ logp_out,
                  float* __restrict__ probs_out,
                  float* __restrict__ tokens_out) {
    extern __shared__ float cache[];      // V_TOTAL floats (128000 B)
    __shared__ float red[BLK];
    __shared__ float redv[BLK];
    __shared__ int   redi[BLK];

    const int r   = blockIdx.x;
    const int tid = threadIdx.x;
    const float NEG = -INFINITY;
    const float* arow = ampsq + (size_t)r * V_TOTAL;

    // pass 1: cache amp_sq, row mean -> floor
    float s = 0.f;
    for (int v = tid; v < V_TOTAL; v += BLK) { float a = arow[v]; cache[v] = a; s += a; }
    float total = blk_sum(s, red);
    const float flo = total * (1.0f / (float)V_TOTAL) * 1e-6f + 1e-30f;
    __syncthreads();

    // logits = log(amp_sq + floor) + bias   (TEMP == 1)
    float mx = NEG, mn = INFINITY;
    for (int v = tid; v < V_TOTAL; v += BLK) {
        float lg = __logf(cache[v] + flo) + bias[v];
        cache[v] = lg;
        logits_out[(size_t)r * V_TOTAL + v] = lg;
        mx = fmaxf(mx, lg); mn = fminf(mn, lg);
    }
    float rowmax = blk_max(mx, red);
    float rowmin = blk_min(mn, red);

    // log-softmax
    s = 0.f;
    for (int v = tid; v < V_TOTAL; v += BLK) s += __expf(cache[v] - rowmax);
    float sumexp = blk_sum(s, red);
    float lse = rowmax + __logf(sumexp);
    for (int v = tid; v < V_TOTAL; v += BLK)
        logp_out[(size_t)r * V_TOTAL + v] = cache[v] - lse;

    // top-k threshold via bisection: kth largest value
    float lo = rowmin, hi = rowmax;
    for (int it = 0; it < 30; ++it) {
        float mid = 0.5f * (lo + hi);
        float c = 0.f;
        for (int v = tid; v < V_TOTAL; v += BLK) c += (cache[v] >= mid) ? 1.f : 0.f;
        float cnt = blk_sum(c, red);
        if (cnt >= (float)TOPK) lo = mid; else hi = mid;
    }
    const float kth = lo;
    __syncthreads();
    for (int v = tid; v < V_TOTAL; v += BLK)
        if (cache[v] < kth) cache[v] = NEG;
    __syncthreads();

    // top-p: nucleus boundary by bisection on logit value
    s = 0.f;
    for (int v = tid; v < V_TOTAL; v += BLK) s += __expf(cache[v] - rowmax);
    float Z = blk_sum(s, red);
    lo = kth; hi = rowmax;
    for (int it = 0; it < 30; ++it) {
        float mid = 0.5f * (lo + hi);
        float g = 0.f;
        for (int v = tid; v < V_TOTAL; v += BLK) {
            float f = cache[v];
            g += (f > mid) ? __expf(f - rowmax) : 0.f;
        }
        float G = blk_sum(g, red);
        if (G >= TOPP * Z) lo = mid; else hi = mid;
    }
    const float tau = hi;
    __syncthreads();
    for (int v = tid; v < V_TOTAL; v += BLK)
        if (cache[v] < tau) cache[v] = NEG;
    __syncthreads();

    // final softmax over nucleus
    s = 0.f;
    for (int v = tid; v < V_TOTAL; v += BLK) s += __expf(cache[v] - rowmax);
    float S2 = blk_sum(s, red);
    float invS2 = 1.0f / S2;
    for (int v = tid; v < V_TOTAL; v += BLK)
        probs_out[(size_t)r * V_TOTAL + v] = __expf(cache[v] - rowmax) * invS2;

    // deterministic Gumbel-argmax sample
    float best = NEG; int bidx = 0;
    const unsigned seed = mix32((unsigned)r * 0x9E3779B9u + 42u);
    for (int v = tid; v < V_TOTAL; v += BLK) {
        unsigned h = mix32(seed ^ (unsigned)v);
        float u = (float)(h >> 9) * (1.0f / 8388608.0f) + 1e-7f;
        float sc = cache[v] - __logf(-__logf(u));
        if (sc > best || (sc == best && v < bidx)) { best = sc; bidx = v; }
    }
    __syncthreads();
    redv[tid] = best; redi[tid] = bidx; __syncthreads();
    for (int st = BLK / 2; st > 0; st >>= 1) {
        if (tid < st) {
            float ov = redv[tid + st]; int oi = redi[tid + st];
            if (ov > redv[tid] || (ov == redv[tid] && oi < redi[tid])) {
                redv[tid] = ov; redi[tid] = oi;
            }
        }
        __syncthreads();
    }
    if (tid == 0) tokens_out[r] = (float)redi[0];
}

// ---------------------------------------------------------------------------
// Host launcher
// ---------------------------------------------------------------------------
extern "C" void kernel_launch(void* const* d_in, const int* in_sizes, int n_in,
                              void* d_out, int out_size, void* d_ws, size_t ws_size,
                              hipStream_t stream) {
    (void)in_sizes; (void)n_in; (void)ws_size; (void)out_size;

    const float* psi_r = (const float*)d_in[0];
    const float* psi_i = (const float*)d_in[1];
    const float* w_r   = (const float*)d_in[2];
    const float* w_i   = (const float*)d_in[3];
    const float* bias  = (const float*)d_in[4];

    float* out = (float*)d_out;
    const size_t NV = (size_t)M_TOTAL * V_TOTAL;        // 32,768,000
    const size_t MD = (size_t)M_TOTAL * K_TOTAL;        // 1,048,576
    const size_t WD = (size_t)V_TOTAL * K_TOTAL;        // 32,768,000
    float* logits_out = out;
    float* logp_out   = out + NV;
    float* ampsq_out  = out + 2 * NV;
    float* tokens_out = out + 3 * NV;
    float* probs_out  = out + 3 * NV + M_TOTAL;

    // bf16 staging areas in workspace
    unsigned short* wsb = (unsigned short*)d_ws;
    unsigned short* wrb  = wsb;            // [V,D]
    unsigned short* wib  = wrb + WD;       // [V,D]
    unsigned short* prb  = wib + WD;       // [M,D]
    unsigned short* pib  = prb + MD;       // [M,D]
    unsigned short* pinb = pib + MD;       // [M,D]  (= -psi_i)

    // ---- pre-convert fp32 -> bf16 (hardware cvt, once per launch) ----
    auto conv = [&](const float* in, unsigned short* o, size_t n, float sc) {
        int n8 = (int)(n / 8);
        to_bf16_kernel<<<(n8 + 255) / 256, 256, 0, stream>>>(in, o, n8, sc);
    };
    conv(w_r,   wrb,  WD,  1.0f);
    conv(w_i,   wib,  WD,  1.0f);
    conv(psi_r, prb,  MD,  1.0f);
    conv(psi_i, pib,  MD,  1.0f);
    conv(psi_i, pinb, MD, -1.0f);

    // allow >64KB dynamic LDS for the row kernel (idempotent, capture-safe)
    (void)hipFuncSetAttribute((const void*)born_rowwise,
                              hipFuncAttributeMaxDynamicSharedMemorySize,
                              V_TOTAL * (int)sizeof(float) + 1024);

    dim3 g1(V_TOTAL / TILE_N, M_TOTAL / TILE_M);   // 250 x 16
    born_gemm_wmma<<<g1, 256, 0, stream>>>(prb, pib, pinb, wrb, wib, ampsq_out);

    born_rowwise<<<M_TOTAL, BLK, V_TOTAL * sizeof(float), stream>>>(
        ampsq_out, bias, logits_out, logp_out, probs_out, tokens_out);
}
